// GraphTransformer_9122510537112
// MI455X (gfx1250) — compile-verified, matrix operand order checked by reading, not run on hardware
//
#include <hip/hip_runtime.h>

// ---- problem constants ----
#define BB 16
#define NN 512
#define DD 512
#define HH 8
#define DH 64
#define FF 2048
#define LL 4
#define KE 16
#define MM (BB * NN)   // 8192 rows

typedef __attribute__((ext_vector_type(16))) __bf16 v16bf;
typedef __attribute__((ext_vector_type(8)))  __bf16 v8bf;
typedef __attribute__((ext_vector_type(8)))  float  v8f;

static __device__ inline __bf16 f2bf(float x) {
    unsigned u = __builtin_bit_cast(unsigned, x);
    u += 0x7FFFu + ((u >> 16) & 1u);                 // round-to-nearest-even
    unsigned short s = (unsigned short)(u >> 16);
    return __builtin_bit_cast(__bf16, s);
}

// Load one 16x32 bf16 WMMA fragment (A layout; B layout symmetric with
// row = column index). `base` points at element (row0, k0) of the tile, rows
// are `stride` elements apart. Lane L reads row L&15, k-chunks at
// ko = (L<16 ? 0 : 8) and ko+16  -> two 16-byte loads.
static __device__ inline v16bf load_frag(const __bf16* base, int stride, int lane) {
    const __bf16* p = base + (size_t)(lane & 15) * stride + ((lane & 16) ? 8 : 0);
    v8bf lo = *(const v8bf*)(p);
    v8bf hi = *(const v8bf*)(p + 16);
    v16bf out;
#pragma unroll
    for (int i = 0; i < 8; ++i) { out[i] = lo[i]; out[i + 8] = hi[i]; }
    return out;
}

static __device__ inline v8f wmma_bf(v16bf a, v16bf b, v8f c) {
    return __builtin_amdgcn_wmma_f32_16x16x32_bf16(false, a, false, b, (short)0, c,
                                                   false, false);
}

static __device__ inline v8f vzero8() {
    v8f z;
#pragma unroll
    for (int i = 0; i < 8; ++i) z[i] = 0.0f;
    return z;
}

// ---------------- embedding + Laplacian PE ----------------
__global__ __launch_bounds__(256) void embed_kernel(
    const int* __restrict__ ids, const float* __restrict__ eig,
    const float* __restrict__ emb, const float* __restrict__ pew,
    const float* __restrict__ peb, float* __restrict__ x, __bf16* __restrict__ xb) {
    int r = blockIdx.x;
    int id = ids[r];
    float e[KE];
#pragma unroll
    for (int k = 0; k < KE; ++k) e[k] = eig[(size_t)r * KE + k];
    for (int d = threadIdx.x; d < DD; d += 256) {
        float acc = emb[(size_t)id * DD + d] + peb[d];
#pragma unroll
        for (int k = 0; k < KE; ++k) acc += e[k] * pew[(size_t)k * DD + d];
        x[(size_t)r * DD + d] = acc;
        xb[(size_t)r * DD + d] = f2bf(acc);
    }
}

// ---------------- f32 W[K,N] -> bf16 W^T[N,K] ----------------
__global__ __launch_bounds__(256) void wconv_kernel(const float* __restrict__ W,
                                                    __bf16* __restrict__ Wt,
                                                    int Kd, int Nc) {
    int o = blockIdx.x * 256 + threadIdx.x;
    if (o >= Kd * Nc) return;
    int k = o % Kd, n = o / Kd;
    Wt[o] = f2bf(W[(size_t)k * Nc + n]);
}

// ---- adj (bool, [b][q][k]) -> transposed f32 bias/scale: [b][k][q] --------
// Value is mask/scale so it can seed the WMMA accumulator directly:
// (Q.K^T + mask/scale) * scale = Q.K^T*scale + mask.
__global__ __launch_bounds__(256) void maskconv_kernel(
    const unsigned char* __restrict__ adj, float* __restrict__ mt) {
    size_t i = (size_t)blockIdx.x * 256 + threadIdx.x;  // over B*N*N (output idx)
    int q = (int)(i & (NN - 1));
    int k = (int)((i >> 9) & (NN - 1));
    int b = (int)(i >> 18);
    mt[i] = adj[((size_t)b << 18) + ((size_t)q << 9) + k] ? -8e30f : 0.0f;
}

// ---------------- WMMA GEMM: C = A(bf16)[M,Kd] * Bt(bf16)[Nc,Kd]^T + bias ----
// EPI bits: 1 = store f32, 2 = store bf16, 4 = relu, 8 = store bf16 as V^T
template <int EPI>
__global__ __launch_bounds__(256) void gemm_kernel(
    const __bf16* __restrict__ A, const __bf16* __restrict__ Bt,
    const float* __restrict__ bias, float* __restrict__ Cf,
    __bf16* __restrict__ Cb, int Nc, int Kd) {
    int lane = threadIdx.x & 31, w = threadIdx.x >> 5;
    int m0 = blockIdx.y * 128 + w * 16;
    int n0 = blockIdx.x * 64;
    v8f acc[4];
#pragma unroll
    for (int t = 0; t < 4; ++t) acc[t] = vzero8();
    const __bf16* arow = A + (size_t)m0 * Kd;
    for (int k0 = 0; k0 < Kd; k0 += 32) {
        v16bf af = load_frag(arow + k0, Kd, lane);
#pragma unroll
        for (int t = 0; t < 4; ++t) {
            v16bf bf_ = load_frag(Bt + (size_t)(n0 + t * 16) * Kd + k0, Kd, lane);
            acc[t] = wmma_bf(af, bf_, acc[t]);
        }
    }
    int rbase = m0 + ((lane & 16) ? 8 : 0);
    int nlane = lane & 15;
#pragma unroll
    for (int t = 0; t < 4; ++t) {
        int col = n0 + t * 16 + nlane;
        float bv = bias[col];
#pragma unroll
        for (int g = 0; g < 8; ++g) {
            int row = rbase + g;
            float v = acc[t][g] + bv;
            if (EPI & 4) v = fmaxf(v, 0.0f);
            if (EPI & 1) Cf[(size_t)row * Nc + col] = v;
            if (EPI & 2) Cb[(size_t)row * Nc + col] = f2bf(v);
            if (EPI & 8) {  // store as V^T: [b, h, dh, n]
                int b = row >> 9, nn2 = row & 511;
                int h = col >> 6, dh = col & 63;
                Cb[(((size_t)(b * HH + h) * DH + dh) << 9) + nn2] = f2bf(v);
            }
        }
    }
}

// ---------------- flash attention (bf16 WMMA, online softmax) ----------------
__global__ __launch_bounds__(256) void attn_kernel(
    const __bf16* __restrict__ qb, const __bf16* __restrict__ kb,
    const __bf16* __restrict__ vtb, const float* __restrict__ mt,
    __bf16* __restrict__ ob) {
    __shared__ __align__(16) __bf16 pbuf[8][16][40];  // per-wave P staging tile
    int lane = threadIdx.x & 31, w = threadIdx.x >> 5;
    int bh = blockIdx.x, b = bh >> 3, h = bh & 7;
    int qbase = blockIdx.y * 128 + w * 16;
    const float scale = 0.125f;
    const float NEG = -1e30f;

    const __bf16* qtile = qb + ((size_t)(b * NN + qbase)) * DD + h * DH;
    v16bf qa0 = load_frag(qtile, DD, lane);
    v16bf qa1 = load_frag(qtile + 32, DD, lane);

    v8f O[4];
    float mrow[8], lrow[8];
#pragma unroll
    for (int t = 0; t < 4; ++t) O[t] = vzero8();
#pragma unroll
    for (int g = 0; g < 8; ++g) { mrow[g] = NEG; lrow[g] = 0.0f; }

    int qr = qbase + ((lane & 16) ? 8 : 0);  // local q row base (within N)
    const float* mtb = mt + ((size_t)b << 18);
    const __bf16* vthead = vtb + (size_t)bh * DH * NN;

    for (int kbse = 0; kbse < NN; kbse += 32) {
        const __bf16* kt0 = kb + ((size_t)(b * NN + kbse)) * DD + h * DH;
        const __bf16* kt1 = kt0 + (size_t)16 * DD;
        int c0 = kbse + (lane & 15), c1 = c0 + 16;
        // seed accumulators with mask/scale bias (one 32B vector load each)
        v8f s0 = *(const v8f*)(mtb + ((size_t)c0 << 9) + qr);
        v8f s1 = *(const v8f*)(mtb + ((size_t)c1 << 9) + qr);
        s0 = wmma_bf(qa0, load_frag(kt0, DD, lane), s0);
        s0 = wmma_bf(qa1, load_frag(kt0 + 32, DD, lane), s0);
        s1 = wmma_bf(qa0, load_frag(kt1, DD, lane), s1);
        s1 = wmma_bf(qa1, load_frag(kt1 + 32, DD, lane), s1);

#pragma unroll
        for (int g = 0; g < 8; ++g) {
            float v0 = s0[g] * scale;
            float v1 = s1[g] * scale;
            float mx = fmaxf(v0, v1);
            mx = fmaxf(mx, __shfl_xor(mx, 1));
            mx = fmaxf(mx, __shfl_xor(mx, 2));
            mx = fmaxf(mx, __shfl_xor(mx, 4));
            mx = fmaxf(mx, __shfl_xor(mx, 8));
            float mnew = fmaxf(mrow[g], mx);
            float alpha = __expf(mrow[g] - mnew);
            float p0 = __expf(v0 - mnew);
            float p1 = __expf(v1 - mnew);
            float ps = p0 + p1;
            ps += __shfl_xor(ps, 1);
            ps += __shfl_xor(ps, 2);
            ps += __shfl_xor(ps, 4);
            ps += __shfl_xor(ps, 8);
            lrow[g] = lrow[g] * alpha + ps;
            mrow[g] = mnew;
#pragma unroll
            for (int t = 0; t < 4; ++t) O[t][g] *= alpha;
            int lr = g + ((lane & 16) ? 8 : 0);
            pbuf[w][lr][lane & 15] = f2bf(p0);
            pbuf[w][lr][(lane & 15) + 16] = f2bf(p1);
        }
        // LDS ops from one wave are in-order (DScnt); compiler inserts waits.
        v16bf pa = load_frag(&pbuf[w][0][0], 40, lane);
#pragma unroll
        for (int t = 0; t < 4; ++t) {
            v16bf vf = load_frag(vthead + (size_t)(t * 16) * NN + kbse, NN, lane);
            O[t] = wmma_bf(pa, vf, O[t]);
        }
    }
#pragma unroll
    for (int g = 0; g < 8; ++g) {
        float inv = 1.0f / lrow[g];
        size_t row = (size_t)(b * NN + qr + g);
#pragma unroll
        for (int t = 0; t < 4; ++t) {
            int col = h * DH + t * 16 + (lane & 15);
            ob[row * DD + col] = f2bf(O[t][g] * inv);
        }
    }
}

// ---------------- fused residual + LayerNorm ----------------
__global__ __launch_bounds__(256) void ln_kernel(
    const float* __restrict__ xin, const float* __restrict__ tadd,
    const float* __restrict__ gam, const float* __restrict__ bet,
    float* __restrict__ xout, __bf16* __restrict__ xbout) {
    __shared__ float red[16];
    int r = blockIdx.x, tid = threadIdx.x;
    size_t base = (size_t)r * DD;
    float v0 = xin[base + tid] + tadd[base + tid];
    float v1 = xin[base + tid + 256] + tadd[base + tid + 256];
    float s = v0 + v1, q = v0 * v0 + v1 * v1;
#pragma unroll
    for (int m = 1; m < 32; m <<= 1) { s += __shfl_xor(s, m); q += __shfl_xor(q, m); }
    int wv = tid >> 5, lane = tid & 31;
    if (lane == 0) { red[wv] = s; red[wv + 8] = q; }
    __syncthreads();
    if (tid == 0) {
        float S = 0.0f, Q = 0.0f;
        for (int i = 0; i < 8; ++i) { S += red[i]; Q += red[i + 8]; }
        red[0] = S; red[8] = Q;
    }
    __syncthreads();
    float mu = red[0] * (1.0f / DD);
    float var = red[8] * (1.0f / DD) - mu * mu;
    float rs = rsqrtf(var + 1e-5f);
    float y0 = (v0 - mu) * rs * gam[tid] + bet[tid];
    float y1 = (v1 - mu) * rs * gam[tid + 256] + bet[tid + 256];
    xout[base + tid] = y0;
    xout[base + tid + 256] = y1;
    xbout[base + tid] = f2bf(y0);
    xbout[base + tid + 256] = f2bf(y1);
}

extern "C" void kernel_launch(void* const* d_in, const int* in_sizes, int n_in,
                              void* d_out, int out_size, void* d_ws, size_t ws_size,
                              hipStream_t stream) {
    (void)in_sizes; (void)n_in; (void)out_size; (void)ws_size;
    const int*           ids  = (const int*)d_in[0];
    const float*         eig  = (const float*)d_in[1];
    const unsigned char* adj  = (const unsigned char*)d_in[2];  // jax bool: 1B/elem
    const float*         emb  = (const float*)d_in[3];
    const float*         pew  = (const float*)d_in[4];
    const float*         peb  = (const float*)d_in[5];

    // workspace carve (256B aligned)
    size_t off = 0;
    char* base = (char*)d_ws;
    auto carve = [&](size_t bytes) -> void* {
        void* p = base + off;
        off += (bytes + 255) & ~(size_t)255;
        return p;
    };
    float*  X    = (float*)carve((size_t)MM * DD * 4);
    float*  TMP  = (float*)carve((size_t)MM * DD * 4);
    float*  MT   = (float*)carve((size_t)BB * NN * NN * 4);  // transposed mask bias
    __bf16* XB   = (__bf16*)carve((size_t)MM * DD * 2);
    __bf16* QB   = (__bf16*)carve((size_t)MM * DD * 2);
    __bf16* KB   = (__bf16*)carve((size_t)MM * DD * 2);
    __bf16* VTB  = (__bf16*)carve((size_t)MM * DD * 2);
    __bf16* OB   = (__bf16*)carve((size_t)MM * DD * 2);
    __bf16* HB   = (__bf16*)carve((size_t)MM * FF * 2);
    __bf16* WQB  = (__bf16*)carve((size_t)DD * DD * 2);
    __bf16* WKB  = (__bf16*)carve((size_t)DD * DD * 2);
    __bf16* WVB  = (__bf16*)carve((size_t)DD * DD * 2);
    __bf16* WOB  = (__bf16*)carve((size_t)DD * DD * 2);
    __bf16* W1B  = (__bf16*)carve((size_t)DD * FF * 2);
    __bf16* W2B  = (__bf16*)carve((size_t)FF * DD * 2);

    embed_kernel<<<MM, 256, 0, stream>>>(ids, eig, emb, pew, peb, X, XB);
    maskconv_kernel<<<(BB * NN * NN) / 256, 256, 0, stream>>>(adj, MT);

    dim3 g512(DD / 64, MM / 128);   // GEMMs with Nc = 512
    dim3 gff(FF / 64, MM / 128);    // GEMM with Nc = 2048
    dim3 gattn(BB * HH, NN / 128);

    for (int i = 0; i < LL; ++i) {
        const float* wq = (const float*)d_in[6]  + (size_t)i * DD * DD;
        const float* bq = (const float*)d_in[7]  + (size_t)i * DD;
        const float* wk = (const float*)d_in[8]  + (size_t)i * DD * DD;
        const float* bk = (const float*)d_in[9]  + (size_t)i * DD;
        const float* wv = (const float*)d_in[10] + (size_t)i * DD * DD;
        const float* bv = (const float*)d_in[11] + (size_t)i * DD;
        const float* wo = (const float*)d_in[12] + (size_t)i * DD * DD;
        const float* bo = (const float*)d_in[13] + (size_t)i * DD;
        const float* g1 = (const float*)d_in[14] + (size_t)i * DD;
        const float* c1 = (const float*)d_in[15] + (size_t)i * DD;
        const float* g2 = (const float*)d_in[16] + (size_t)i * DD;
        const float* c2 = (const float*)d_in[17] + (size_t)i * DD;
        const float* w1 = (const float*)d_in[18] + (size_t)i * DD * FF;
        const float* b1 = (const float*)d_in[19] + (size_t)i * FF;
        const float* w2 = (const float*)d_in[20] + (size_t)i * FF * DD;
        const float* b2 = (const float*)d_in[21] + (size_t)i * DD;

        wconv_kernel<<<(DD * DD) / 256, 256, 0, stream>>>(wq, WQB, DD, DD);
        wconv_kernel<<<(DD * DD) / 256, 256, 0, stream>>>(wk, WKB, DD, DD);
        wconv_kernel<<<(DD * DD) / 256, 256, 0, stream>>>(wv, WVB, DD, DD);
        wconv_kernel<<<(DD * DD) / 256, 256, 0, stream>>>(wo, WOB, DD, DD);
        wconv_kernel<<<(DD * FF) / 256, 256, 0, stream>>>(w1, W1B, DD, FF);
        wconv_kernel<<<(FF * DD) / 256, 256, 0, stream>>>(w2, W2B, FF, DD);

        gemm_kernel<2><<<g512, 256, 0, stream>>>(XB, WQB, bq, nullptr, QB, DD, DD);
        gemm_kernel<2><<<g512, 256, 0, stream>>>(XB, WKB, bk, nullptr, KB, DD, DD);
        gemm_kernel<8><<<g512, 256, 0, stream>>>(XB, WVB, bv, nullptr, VTB, DD, DD);

        attn_kernel<<<gattn, 256, 0, stream>>>(QB, KB, VTB, MT, OB);

        gemm_kernel<1><<<g512, 256, 0, stream>>>(OB, WOB, bo, TMP, nullptr, DD, DD);
        ln_kernel<<<MM, 256, 0, stream>>>(X, TMP, g1, c1, X, XB);

        gemm_kernel<6><<<gff, 256, 0, stream>>>(XB, W1B, b1, nullptr, HB, FF, DD);
        gemm_kernel<1><<<g512, 256, 0, stream>>>(HB, W2B, b2, TMP, nullptr, DD, FF);
        ln_kernel<<<MM, 256, 0, stream>>>(X, TMP, g2, c2, X, XB);
    }

    hipMemcpyAsync(d_out, X, (size_t)MM * DD * 4, hipMemcpyDeviceToDevice, stream);
}
